// LinearProbeHead_55147380081236
// MI455X (gfx1250) — matrix operand order repaired and checked by reading, compiled, not run on hardware
//
#include <hip/hip_runtime.h>
#include <stdint.h>
#include <stddef.h>

// Problem constants from the reference: x[16384,1024], mu[1024,1024], std[1024]
#define N_SAMP 16384
#define KDIM   1024
#define MCLS   1024

#define KCHUNK     128                  // K-extent staged in LDS per buffer
#define NCHUNK     (KDIM / KCHUNK)      // 8
#define ROW_SHORTS 136                  // 128 data + 8 pad -> 272B row stride (bank-conflict free)
#define BUF_SHORTS (64 * ROW_SHORTS)    // one buffer: 64 cols x padded row

typedef __bf16          v16bf  __attribute__((ext_vector_type(16)));
typedef float           v8f    __attribute__((ext_vector_type(8)));
typedef float           f32x8  __attribute__((ext_vector_type(8)));
typedef unsigned int    u32x8  __attribute__((ext_vector_type(8)));

__device__ __forceinline__ unsigned short f32_to_bf16_rne(float f) {
  unsigned int u = __float_as_uint(f);
  u += 0x7FFFu + ((u >> 16) & 1u);   // round to nearest even (prep kernels only)
  return (unsigned short)(u >> 16);
}

// Pack two f32 into one dword of two bf16 (round-to-nearest) with a single v_perm_b32.
__device__ __forceinline__ unsigned pack2_bf16(float f0, float f1) {
  const unsigned u0 = __float_as_uint(f0) + 0x8000u;
  const unsigned u1 = __float_as_uint(f1) + 0x8000u;
  return __builtin_amdgcn_perm(u1, u0, 0x07060302u);
}

__device__ __forceinline__ v16bf pack_A(const f32x8& a0, const f32x8& a1) {
  u32x8 av;
  av[0] = pack2_bf16(a0[0], a0[1]);  av[1] = pack2_bf16(a0[2], a0[3]);
  av[2] = pack2_bf16(a0[4], a0[5]);  av[3] = pack2_bf16(a0[6], a0[7]);
  av[4] = pack2_bf16(a1[0], a1[1]);  av[5] = pack2_bf16(a1[2], a1[3]);
  av[6] = pack2_bf16(a1[4], a1[5]);  av[7] = pack2_bf16(a1[6], a1[7]);
  return __builtin_bit_cast(v16bf, av);
}

// ---------------------------------------------------------------------------
// Kernel 1: per class-row m: row norm of mu, half_mm[m], packed bf16 weights
// ---------------------------------------------------------------------------
__global__ void __launch_bounds__(256)
prep_mu_kernel(const float* __restrict__ mu, const float* __restrict__ stdv,
               unsigned short* __restrict__ Wbf, float* __restrict__ half_mm) {
  const int m   = blockIdx.x;
  const int tid = threadIdx.x;
  __shared__ float r1[256];
  __shared__ float r2[256];
  float s1 = 0.f, s2 = 0.f;
  #pragma unroll
  for (int i = 0; i < KDIM / 256; ++i) {
    const int k = tid + i * 256;
    const float v  = mu[(size_t)m * KDIM + k];
    const float is = 1.0f / stdv[k];
    s1 += v * v;
    s2 += v * v * is;
  }
  r1[tid] = s1; r2[tid] = s2;
  __syncthreads();
  for (int off = 128; off > 0; off >>= 1) {
    if (tid < off) { r1[tid] += r1[tid + off]; r2[tid] += r2[tid + off]; }
    __syncthreads();
  }
  __shared__ float s_invnorm;
  if (tid == 0) {
    const float sum1 = r1[0], sum2 = r2[0];
    s_invnorm  = rsqrtf(sum1);
    half_mm[m] = 0.5f * sum2 / sum1;
  }
  __syncthreads();
  const float invn = s_invnorm;
  #pragma unroll
  for (int i = 0; i < KDIM / 256; ++i) {
    const int k = tid + i * 256;
    const float w = mu[(size_t)m * KDIM + k] * invn * (1.0f / stdv[k]);
    Wbf[(size_t)m * KDIM + k] = f32_to_bf16_rne(w);
  }
}

// ---------------------------------------------------------------------------
// Kernel 2: half_xx[n] = 0.5 * sum_k x[n,k]^2 / std[k]
// ---------------------------------------------------------------------------
__global__ void __launch_bounds__(256)
prep_x_kernel(const float* __restrict__ x, const float* __restrict__ stdv,
              float* __restrict__ half_xx) {
  const int n   = blockIdx.x;
  const int tid = threadIdx.x;
  __shared__ float red[256];
  float s = 0.f;
  #pragma unroll
  for (int i = 0; i < KDIM / 256; ++i) {
    const int k = tid + i * 256;
    const float v = x[(size_t)n * KDIM + k];
    s += v * v * (1.0f / stdv[k]);
  }
  red[tid] = s;
  __syncthreads();
  for (int off = 128; off > 0; off >>= 1) {
    if (tid < off) red[tid] += red[tid + off];
    __syncthreads();
  }
  if (tid == 0) half_xx[n] = 0.5f * red[0];
}

// ---------------------------------------------------------------------------
// Kernel 3: WMMA GEMM  out[n,m] = cross[n,m] - half_xx[n] - half_mm[m]
//   - W staged in LDS: global_load_async_to_lds_b128, 2 chunks in flight.
//   - A: depth-2 register pipeline over global loads (wait loadcnt<=4).
//   - B: grouped ds loads per k-step; A-conversion VALU covers LDS latency.
//   - __builtin_amdgcn_sched_barrier(0) fences pin the hand schedule.
// ---------------------------------------------------------------------------
__global__ void __launch_bounds__(256)
probe_gemm_kernel(const float* __restrict__ x, const unsigned short* __restrict__ Wbf,
                  const float* __restrict__ half_xx, const float* __restrict__ half_mm,
                  float* __restrict__ out) {
  __shared__ unsigned short Bsm[2 * BUF_SHORTS];   // ~34 KB

  const int tid    = threadIdx.x;
  const int lane   = tid & 31;
  const int wave   = tid >> 5;          // 0..7
  const int lane16 = lane & 15;
  const int half   = lane >> 4;         // 0/1

  const int rowBase = blockIdx.y * 128 + wave * 16;  // sample rows (WMMA M)
  const int colBase = blockIdx.x * 64;               // class cols  (WMMA N)

  const float* __restrict__ xrow = x + (size_t)(rowBase + lane16) * KDIM;

  // --- async staging map: thread -> (B row, 16-byte slice); 4 rows/thread/chunk
  const int      crow = tid >> 4;                     // 0..15
  const unsigned ckb  = (unsigned)((tid & 15) * 16);  // byte slice within 256B row
  const unsigned ldsBase = (unsigned)(size_t)&Bsm[0]; // low 32b of flat == LDS offset

  auto issue_chunk = [&](int buf, int kt) {
    #pragma unroll
    for (int p = 0; p < 4; ++p) {
      const int row = crow + p * 16;  // col within block tile
      const unsigned goff = (unsigned)(((colBase + row) * KDIM + kt) * 2) + ckb;
      const unsigned loff = ldsBase + (unsigned)(buf * BUF_SHORTS + row * ROW_SHORTS) * 2 + ckb;
      asm volatile("global_load_async_to_lds_b128 %0, %1, %2"
                   :: "v"(loff), "v"(goff), "s"(Wbf) : "memory");
    }
  };

  v8f acc0 = {}, acc1 = {}, acc2 = {}, acc3 = {};

  issue_chunk(0, 0);
  issue_chunk(1, KCHUNK);

  // A pipeline prologue: stage k-steps 0 (even) and 1 (odd)
  f32x8 a0e = *(const f32x8*)(xrow + 0 + half * 16);
  f32x8 a1e = *(const f32x8*)(xrow + 8 + half * 16);
  f32x8 a0o = *(const f32x8*)(xrow + 32 + half * 16);
  f32x8 a1o = *(const f32x8*)(xrow + 40 + half * 16);

  asm volatile("s_wait_asynccnt 0x4" ::: "memory");   // chunk 0 complete
  __syncthreads();

  for (int c = 0; c < NCHUNK; ++c) {
    const int buf = c & 1;
    const unsigned short* __restrict__ Bb =
        &Bsm[buf * BUF_SHORTS + lane16 * ROW_SHORTS + half * 16];

    #pragma unroll
    for (int kk = 0; kk < KCHUNK; kk += 32) {
      const int  s  = kk >> 5;          // 0..3 (steps/chunk is even -> parity stable)
      const bool ev = (s & 1) == 0;

      // ---- S1: issue this step's B tiles (8x ds_load_b128)
      const u32x8 b0 = *(const u32x8*)(Bb + 0 * 16 * ROW_SHORTS + kk);
      const u32x8 b1 = *(const u32x8*)(Bb + 1 * 16 * ROW_SHORTS + kk);
      const u32x8 b2 = *(const u32x8*)(Bb + 2 * 16 * ROW_SHORTS + kk);
      const u32x8 b3 = *(const u32x8*)(Bb + 3 * 16 * ROW_SHORTS + kk);
      __builtin_amdgcn_sched_barrier(0);

      // ---- S2: convert A stage (loaded 2 steps ago), then refill it (step s+2)
      const v16bf A = pack_A(ev ? a0e : a0o, ev ? a1e : a1o);
      const int kpre  = c * KCHUNK + kk + 64;
      const int ksafe = (kpre < KDIM) ? kpre : 0;
      if (ev) {
        a0e = *(const f32x8*)(xrow + ksafe + half * 16);
        a1e = *(const f32x8*)(xrow + ksafe + 8 + half * 16);
      } else {
        a0o = *(const f32x8*)(xrow + ksafe + half * 16);
        a1o = *(const f32x8*)(xrow + ksafe + 8 + half * 16);
      }
      __builtin_amdgcn_sched_barrier(0);

      // ---- S3: 4 back-to-back WMMAs
      acc0 = __builtin_amdgcn_wmma_f32_16x16x32_bf16(false, A, false,
               __builtin_bit_cast(v16bf, b0), (short)0, acc0, false, false);
      acc1 = __builtin_amdgcn_wmma_f32_16x16x32_bf16(false, A, false,
               __builtin_bit_cast(v16bf, b1), (short)0, acc1, false, false);
      acc2 = __builtin_amdgcn_wmma_f32_16x16x32_bf16(false, A, false,
               __builtin_bit_cast(v16bf, b2), (short)0, acc2, false, false);
      acc3 = __builtin_amdgcn_wmma_f32_16x16x32_bf16(false, A, false,
               __builtin_bit_cast(v16bf, b3), (short)0, acc3, false, false);
      __builtin_amdgcn_sched_barrier(0);
    }

    __syncthreads();                      // all waves done reading buffer `buf`
    if (c + 1 < NCHUNK) {
      if (c + 2 < NCHUNK) {
        issue_chunk(buf, (c + 2) * KCHUNK);               // refill freed buffer
        asm volatile("s_wait_asynccnt 0x4" ::: "memory"); // chunk c+1 complete
      } else {
        asm volatile("s_wait_asynccnt 0x0" ::: "memory"); // last chunk complete
      }
      __syncthreads();
    }
  }

  // --- Epilogue. D layout: lanes 0-15: VGPR r -> (M=r, N=lane); lanes 16-31: (M=r+8, N=lane-16)
  float hx[8];
  #pragma unroll
  for (int r = 0; r < 8; ++r) hx[r] = half_xx[rowBase + half * 8 + r];

  #pragma unroll
  for (int t = 0; t < 4; ++t) {
    const int col = colBase + t * 16 + lane16;
    const float hm = half_mm[col];
    const v8f a = (t == 0) ? acc0 : (t == 1) ? acc1 : (t == 2) ? acc2 : acc3;
    #pragma unroll
    for (int r = 0; r < 8; ++r) {
      const int row = rowBase + half * 8 + r;
      // streamed once -> non-temporal store keeps x resident in L2
      __builtin_nontemporal_store(a[r] - hx[r] - hm, &out[(size_t)row * MCLS + col]);
    }
  }
}

// ---------------------------------------------------------------------------
extern "C" void kernel_launch(void* const* d_in, const int* in_sizes, int n_in,
                              void* d_out, int out_size, void* d_ws, size_t ws_size,
                              hipStream_t stream) {
  (void)in_sizes; (void)n_in; (void)out_size; (void)ws_size;
  const float* x    = (const float*)d_in[0];
  const float* mu   = (const float*)d_in[1];
  const float* stdv = (const float*)d_in[2];
  float* out = (float*)d_out;

  // Workspace: [half_mm: 1024 f32][half_xx: 16384 f32][Wbf16: 1024*1024 u16] (~2.07 MB)
  float* half_mm = (float*)d_ws;
  float* half_xx = half_mm + MCLS;
  unsigned short* Wbf = (unsigned short*)(half_xx + N_SAMP);

  prep_mu_kernel<<<MCLS, 256, 0, stream>>>(mu, stdv, Wbf, half_mm);
  prep_x_kernel<<<N_SAMP, 256, 0, stream>>>(x, stdv, half_xx);

  dim3 grid(MCLS / 64, N_SAMP / 128);  // (16, 128) blocks of 128x64 output
  probe_gemm_kernel<<<grid, 256, 0, stream>>>(x, Wbf, half_xx, half_mm, out);
}
